// SelfAttention_1116691497148
// MI455X (gfx1250) — matrix operand order repaired and checked by reading, not compiled, not run on hardware
//
#include <hip/hip_runtime.h>
#include <hip/hip_bf16.h>

typedef _Float16 half_t;
typedef __attribute__((ext_vector_type(16))) _Float16 v16h;
typedef __attribute__((ext_vector_type(8)))  _Float16 v8h;
typedef __attribute__((ext_vector_type(4)))  _Float16 v4h;
typedef __attribute__((ext_vector_type(8)))  float    v8f;

// Problem constants (match reference)
constexpr int B_   = 4;
constexpr int S_   = 1024;
constexpr int DIM_ = 2048;
constexpr int HQ_  = 16;
constexpr int HKV_ = 4;
constexpr int HD_  = 128;
constexpr float kScale = 0.08838834764831845f;   // 1/sqrt(128)

// ds_swizzle lane exchange (group-of-32 mode: and=0x1F, or=0, xor=imm>>10)
#define SWZ_F(v, imm) __builtin_bit_cast(float, \
    __builtin_amdgcn_ds_swizzle(__builtin_bit_cast(int, (float)(v)), (imm)))

__device__ inline float rmax16(float v) {   // max over 16-lane group
  v = fmaxf(v, SWZ_F(v, 0x041F));
  v = fmaxf(v, SWZ_F(v, 0x081F));
  v = fmaxf(v, SWZ_F(v, 0x101F));
  v = fmaxf(v, SWZ_F(v, 0x201F));
  return v;
}
__device__ inline float rsum16(float v) {   // sum over 16-lane group
  v += SWZ_F(v, 0x041F);
  v += SWZ_F(v, 0x081F);
  v += SWZ_F(v, 0x101F);
  v += SWZ_F(v, 0x201F);
  return v;
}

// A-matrix 16x32 f16 fragment (ISA 7.12.2): lane<16 row=lane, K={0..7,16..23};
// lane>=16 same row, K={8..15,24..31}.
__device__ inline v16h load_fragA(const half_t* p, int ld) {
  const int lane = threadIdx.x & 31;
  const int r  = lane & 15;
  const int kb = (lane >> 4) << 3;           // 0 or 8
  const half_t* q = p + (size_t)r * ld + kb;
  v8h lo = *reinterpret_cast<const v8h*>(q);       // K kb..kb+7
  v8h hi = *reinterpret_cast<const v8h*>(q + 16);  // K kb+16..kb+23
  return __builtin_shufflevector(lo, hi, 0,1,2,3,4,5,6,7,8,9,10,11,12,13,14,15);
}

// B-matrix 32x16 f16 fragment: lane<16 col=lane, K={0..7,16..23};
// lane>=16 same col, K={8..15,24..31} (matches A fragment k order).
__device__ inline v16h load_fragB(const half_t* p, int ld) {
  const int lane = threadIdx.x & 31;
  const int c  = lane & 15;
  const int kb = (lane >> 4) << 3;           // 0 or 8
  const half_t* q = p + (size_t)c * ld + kb;
  v8h lo = *reinterpret_cast<const v8h*>(q);
  v8h hi = *reinterpret_cast<const v8h*>(q + 16);
  return __builtin_shufflevector(lo, hi, 0,1,2,3,4,5,6,7,8,9,10,11,12,13,14,15);
}

__device__ inline v8f wmma32(v16h a, v16h b, v8f c) {
  return __builtin_amdgcn_wmma_f32_16x16x32_f16(
      /*neg_a=*/false, a, /*neg_b=*/false, b,
      /*c_mod=*/(short)0, c, /*reuse_a=*/false, /*reuse_b=*/false);
}

// ---------------------------------------------------------------------------
// fp32 -> fp16 conversion (vectorized by 4)
// ---------------------------------------------------------------------------
__global__ void f32_to_f16(const float* __restrict__ src,
                           half_t* __restrict__ dst, int n4) {
  int i = blockIdx.x * blockDim.x + threadIdx.x;
  if (i < n4) {
    float4 v = reinterpret_cast<const float4*>(src)[i];
    v4h o = { (half_t)v.x, (half_t)v.y, (half_t)v.z, (half_t)v.w };
    *reinterpret_cast<v4h*>(dst + (size_t)i * 4) = o;
  }
}

// ---------------------------------------------------------------------------
// WMMA GEMM: C[M x N] = A[M x K] * W[N x K]^T  (both operands K-contiguous).
// One wave computes a 64x64 tile. K is processed 64-deep per iteration with
// a two-stage register ping-pong so a full fragment set (4 A + 4 B) is always
// in flight beneath the 16 WMMAs of the other stage. MODE: 0=Q(+RoPE),
// 1=K(+RoPE), 2=V (store transposed (b,h,d,s)), 3=plain f32 out (O-proj).
// ---------------------------------------------------------------------------
template<int MODE>
__global__ __launch_bounds__(32)
void gemm_wmma(const half_t* __restrict__ A, const half_t* __restrict__ W,
               int Kdim, const float* __restrict__ freqs,
               half_t* __restrict__ outH, float* __restrict__ outF) {
  const int lane = threadIdx.x;
  const int m0 = blockIdx.y * 64;
  const int n0 = blockIdx.x * 64;
  const half_t* Abase = A + (size_t)m0 * Kdim;
  const half_t* Wbase = W + (size_t)n0 * Kdim;

  v8f acc[4][4];
#pragma unroll
  for (int mi = 0; mi < 4; ++mi)
#pragma unroll
    for (int ni = 0; ni < 4; ++ni) acc[mi][ni] = v8f{};

  v16h a0[4], b0[4], a1[4], b1[4];
#pragma unroll
  for (int mi = 0; mi < 4; ++mi)
    a0[mi] = load_fragA(Abase + (size_t)mi * 16 * Kdim, Kdim);
#pragma unroll
  for (int ni = 0; ni < 4; ++ni)
    b0[ni] = load_fragB(Wbase + (size_t)ni * 16 * Kdim, Kdim);

  for (int k0 = 0; k0 < Kdim; k0 += 64) {     // Kdim is a multiple of 64
    __builtin_prefetch(Wbase + k0 + 512, 0, 1);
    // stage 1 fragments (k0+32) load under stage 0 WMMAs
#pragma unroll
    for (int mi = 0; mi < 4; ++mi)
      a1[mi] = load_fragA(Abase + (size_t)mi * 16 * Kdim + k0 + 32, Kdim);
#pragma unroll
    for (int ni = 0; ni < 4; ++ni)
      b1[ni] = load_fragB(Wbase + (size_t)ni * 16 * Kdim + k0 + 32, Kdim);
#pragma unroll
    for (int ni = 0; ni < 4; ++ni)
#pragma unroll
      for (int mi = 0; mi < 4; ++mi)
        acc[mi][ni] = wmma32(a0[mi], b0[ni], acc[mi][ni]);
    if (k0 + 64 < Kdim) {                     // uniform branch
#pragma unroll
      for (int mi = 0; mi < 4; ++mi)
        a0[mi] = load_fragA(Abase + (size_t)mi * 16 * Kdim + k0 + 64, Kdim);
#pragma unroll
      for (int ni = 0; ni < 4; ++ni)
        b0[ni] = load_fragB(Wbase + (size_t)ni * 16 * Kdim + k0 + 64, Kdim);
    }
#pragma unroll
    for (int ni = 0; ni < 4; ++ni)
#pragma unroll
      for (int mi = 0; mi < 4; ++mi)
        acc[mi][ni] = wmma32(a1[mi], b1[ni], acc[mi][ni]);
  }

  const int lh = lane >> 4;        // which half of the wave
  const int nl = lane & 15;        // column within 16-wide tile
#pragma unroll
  for (int mi = 0; mi < 4; ++mi) {
#pragma unroll
    for (int ni = 0; ni < 4; ++ni) {
#pragma unroll
      for (int j = 0; j < 8; ++j) {
        const int gm = m0 + mi * 16 + j + (lh << 3);   // flattened b*S+s
        const int n  = n0 + ni * 16 + nl;              // output feature
        const int b  = gm >> 10;                       // S_ = 1024
        const int s  = gm & (S_ - 1);
        float val = acc[mi][ni][j];
        if constexpr (MODE == 0 || MODE == 1) {
          // Llama RoPE on adjacent (even,odd) pairs: partner is lane^1.
          const int d = n & (HD_ - 1);
          const float ang = freqs[(size_t)s * (HD_ / 2) + (d >> 1)];
          float cs, sn;
          sincosf(ang, &sn, &cs);
          const float other = SWZ_F(val, 0x041F);      // lane ^ 1
          val = ((n & 1) == 0) ? (val * cs - other * sn)
                               : (val * cs + other * sn);
        }
        if constexpr (MODE == 0) {
          const int h = n >> 7, d = n & 127;
          outH[(((size_t)b * HQ_ + h) * S_ + s) * HD_ + d] = (half_t)val;
        } else if constexpr (MODE == 1) {
          const int h = n >> 7, d = n & 127;
          outH[(((size_t)b * HKV_ + h) * S_ + s) * HD_ + d] = (half_t)val;
        } else if constexpr (MODE == 2) {
          const int h = n >> 7, d = n & 127;   // V stored transposed (d,s)
          outH[(((size_t)b * HKV_ + h) * HD_ + d) * S_ + s] = (half_t)val;
        } else {
          outF[(size_t)gm * DIM_ + n] = val;
        }
      }
    }
  }
}

// ---------------------------------------------------------------------------
// Flash attention, one wave per (b, q-head, 16-query block).
// Key blocks of 32 -> Q.K^T (8 WMMA) -> online softmax -> P via LDS -> P.V
// (8 WMMA into 8 f32 accumulators covering HD=128).
// ---------------------------------------------------------------------------
__global__ __launch_bounds__(32)
void flash_attn(const half_t* __restrict__ Qh, const half_t* __restrict__ Kh,
                const half_t* __restrict__ Vt, half_t* __restrict__ Oh) {
  const int lane = threadIdx.x;
  const int qb = blockIdx.x;        // 16-query block
  const int hq = blockIdx.y;
  const int b  = blockIdx.z;
  const int hkv = hq >> 2;          // G = 4

  const half_t* Qp = Qh + (((size_t)b * HQ_ + hq) * S_ + (size_t)qb * 16) * HD_;
  const half_t* Kbase = Kh + ((size_t)b * HKV_ + hkv) * (size_t)S_ * HD_;
  const half_t* Vbase = Vt + ((size_t)b * HKV_ + hkv) * (size_t)HD_ * S_;

  v16h qf[4];
#pragma unroll
  for (int ks = 0; ks < 4; ++ks) qf[ks] = load_fragA(Qp + ks * 32, HD_);

  v8f oacc[8];
#pragma unroll
  for (int dt = 0; dt < 8; ++dt) oacc[dt] = v8f{};
  float mi[8], li[8];
#pragma unroll
  for (int j = 0; j < 8; ++j) { mi[j] = -1e30f; li[j] = 0.0f; }

  __shared__ __align__(16) half_t Pt[16 * 32];

  const int lh = lane >> 4;
  const int nl = lane & 15;
  const int smax = qb * 16 + 15;
  const int nkb = (smax >> 5) + 1;          // causal: only blocks <= diagonal

  for (int kb = 0; kb < nkb; ++kb) {
    const int t0 = kb * 32;
    const half_t* Kp = Kbase + (size_t)t0 * HD_;
    __builtin_prefetch(Kbase + (size_t)(t0 + 32) * HD_, 0, 0);

    v8f s0 = v8f{}, s1 = v8f{};
#pragma unroll
    for (int ks = 0; ks < 4; ++ks) {
      v16h b0 = load_fragB(Kp + ks * 32, HD_);
      v16h b1 = load_fragB(Kp + (size_t)16 * HD_ + ks * 32, HD_);
      s0 = wmma32(qf[ks], b0, s0);
      s1 = wmma32(qf[ks], b1, s1);
    }

    float pf[8];
#pragma unroll
    for (int j = 0; j < 8; ++j) {
      const int srow = qb * 16 + j + (lh << 3);
      float v0 = s0[j] * kScale;
      float v1 = s1[j] * kScale;
      if (t0 + nl > srow)      v0 = -1e9f;   // causal mask
      if (t0 + 16 + nl > srow) v1 = -1e9f;
      const float rm   = rmax16(fmaxf(v0, v1));
      const float mnew = fmaxf(mi[j], rm);
      const float sc   = __expf(mi[j] - mnew);
      const float e0   = __expf(v0 - mnew);
      const float e1   = __expf(v1 - mnew);
      li[j] = li[j] * sc + rsum16(e0 + e1);
      mi[j] = mnew;
      pf[j] = sc;
      const int row = j + (lh << 3);
      Pt[row * 32 + nl]      = (half_t)e0;
      Pt[row * 32 + 16 + nl] = (half_t)e1;
    }
    __syncthreads();                         // make P visible cross-lane
    v16h pa = load_fragA(Pt, 32);            // P as 16x32 A-fragment

#pragma unroll
    for (int dt = 0; dt < 8; ++dt)
#pragma unroll
      for (int j = 0; j < 8; ++j) oacc[dt][j] *= pf[j];

    const half_t* Vp = Vbase + t0;
#pragma unroll
    for (int dt = 0; dt < 8; ++dt) {
      v16h bv = load_fragB(Vp + (size_t)dt * 16 * S_, S_);
      oacc[dt] = wmma32(pa, bv, oacc[dt]);
    }
    __syncthreads();                         // protect Pt for next iteration
  }

#pragma unroll
  for (int dt = 0; dt < 8; ++dt) {
#pragma unroll
    for (int j = 0; j < 8; ++j) {
      const int srow = qb * 16 + j + (lh << 3);
      const int d = dt * 16 + nl;
      const float val = oacc[dt][j] / li[j];
      Oh[((size_t)b * S_ + srow) * DIM_ + hq * HD_ + d] = (half_t)val;
    }
  }
}

// ---------------------------------------------------------------------------
// Host launch
// ---------------------------------------------------------------------------
extern "C" void kernel_launch(void* const* d_in, const int* in_sizes, int n_in,
                              void* d_out, int out_size, void* d_ws, size_t ws_size,
                              hipStream_t stream) {
  (void)in_sizes; (void)n_in; (void)out_size; (void)ws_size;
  const float* x     = (const float*)d_in[0];
  // d_in[1] = start_pos (0, unused); d_in[3] = mask (causal, computed inline)
  const float* freqs = (const float*)d_in[2];
  const float* wq    = (const float*)d_in[4];
  const float* wk    = (const float*)d_in[5];
  const float* wv    = (const float*)d_in[6];
  const float* wo    = (const float*)d_in[7];
  float* out = (float*)d_out;

  char* ws = (char*)d_ws;
  size_t off = 0;
  auto carve = [&](size_t elems) {
    half_t* p = (half_t*)(ws + off);
    off += ((elems * sizeof(half_t) + 255) / 256) * 256;
    return p;
  };
  half_t* xh  = carve((size_t)B_ * S_ * DIM_);
  half_t* wqh = carve((size_t)HQ_  * HD_ * DIM_);
  half_t* wkh = carve((size_t)HKV_ * HD_ * DIM_);
  half_t* wvh = carve((size_t)HKV_ * HD_ * DIM_);
  half_t* woh = carve((size_t)DIM_ * DIM_);
  half_t* Qh  = carve((size_t)B_ * HQ_  * S_ * HD_);
  half_t* Kh  = carve((size_t)B_ * HKV_ * S_ * HD_);
  half_t* Vth = carve((size_t)B_ * HKV_ * HD_ * S_);
  half_t* Oh  = carve((size_t)B_ * S_ * DIM_);

  auto conv = [&](const float* s, half_t* d, size_t n) {
    int n4 = (int)(n / 4);
    f32_to_f16<<<(n4 + 255) / 256, 256, 0, stream>>>(s, d, n4);
  };
  conv(x,  xh,  (size_t)B_ * S_ * DIM_);
  conv(wq, wqh, (size_t)HQ_  * HD_ * DIM_);
  conv(wk, wkh, (size_t)HKV_ * HD_ * DIM_);
  conv(wv, wvh, (size_t)HKV_ * HD_ * DIM_);
  conv(wo, woh, (size_t)DIM_ * DIM_);

  const dim3 wave(32);
  const int Mrows = B_ * S_;                       // 4096
  gemm_wmma<0><<<dim3((HQ_  * HD_) / 64, Mrows / 64), wave, 0, stream>>>(
      xh, wqh, DIM_, freqs, Qh, nullptr);
  gemm_wmma<1><<<dim3((HKV_ * HD_) / 64, Mrows / 64), wave, 0, stream>>>(
      xh, wkh, DIM_, freqs, Kh, nullptr);
  gemm_wmma<2><<<dim3((HKV_ * HD_) / 64, Mrows / 64), wave, 0, stream>>>(
      xh, wvh, DIM_, nullptr, Vth, nullptr);

  flash_attn<<<dim3(S_ / 16, HQ_, B_), wave, 0, stream>>>(Qh, Kh, Vth, Oh);

  gemm_wmma<3><<<dim3(DIM_ / 64, Mrows / 64), wave, 0, stream>>>(
      Oh, woh, DIM_, nullptr, nullptr, out);
}
